// VectorQuantizerEMA_34737695490440
// MI455X (gfx1250) — matrix operand order repaired and checked by reading, not compiled, not run on hardware
//
#include <hip/hip_runtime.h>
#include <hip/hip_bf16.h>
#include <math.h>

// ---------------- problem constants ----------------
#define DIM     64
#define NE      512
#define NCODES  (64*64*64)            // 262144 rows
#define NTILES  (NE/16)               // 32 column tiles
#define ROWS_PER_WG 128               // 8 waves x 16 rows
#define ROW_TILES   (NCODES/ROWS_PER_WG)  // 2048
#define NUM_WG  256
#define THREADS 256

// output layout: [loss, perplexity, new_w(64*512), new_cluster_size(512), new_embed_avg(64*512)]
#define OUT_W   2
#define OUT_CS  (2 + DIM*NE)          // 32770
#define OUT_EA  (OUT_CS + NE)         // 33282

typedef __attribute__((ext_vector_type(16))) __bf16        v16bf;
typedef __attribute__((ext_vector_type(16))) unsigned short v16u;
typedef __attribute__((ext_vector_type(8)))  float          v8f;

__device__ __forceinline__ unsigned short f32_to_bf16(float f) {
  // round-to-nearest-even f32 -> bf16
  unsigned int u = __float_as_uint(f);
  u = (u + 0x7FFFu + ((u >> 16) & 1u)) >> 16;
  return (unsigned short)u;
}

// ---------------- init: zero the accumulators in d_out ----------------
__global__ void vq_init(float* __restrict__ out) {
  int i = blockIdx.x * blockDim.x + threadIdx.x;
  if (i == 0) { out[0] = 0.f; out[1] = 0.f; }
  if (i < NE)      out[OUT_CS + i] = 0.f;
  if (i < DIM*NE)  out[OUT_EA + i] = 0.f;
}

// ---------------- main: WMMA scores + argmax + segment sums ----------------
__launch_bounds__(THREADS, 1)
__global__ void vq_main(const float* __restrict__ x,
                        const float* __restrict__ w,
                        float* __restrict__ out) {
  // LDS budget (CDNA5: 320KB/WGP):
  __shared__ float lds_embed[DIM*NE];                               // 128 KB partial embed_sum
  __shared__ alignas(32) unsigned short lds_B[2*NTILES*32*16];      //  64 KB bf16 B fragments
  __shared__ float lds_ww[NE];                                      //   2 KB -0.5*||w_j||^2
  __shared__ float lds_counts[NE];                                  //   2 KB histogram
  __shared__ float lds_xx[8][16];                                   // per-wave row norms
  __shared__ int   lds_win[8][16];                                  // per-wave row winners

  const int tid  = threadIdx.x;
  const int lane = tid & 31;
  const int wave = tid >> 5;
  const int h    = (lane >> 4) & 1;   // half-wave
  const int lm   = lane & 15;

  // ---- fill phase (once per WG) ----
  for (int i = tid; i < DIM*NE; i += THREADS) lds_embed[i] = 0.f;
  for (int i = tid; i < NE; i += THREADS) {
    lds_counts[i] = 0.f;
    float s = 0.f;
    #pragma unroll
    for (int d = 0; d < DIM; ++d) { float v = w[d*NE + i]; s += v*v; }
    lds_ww[i] = -0.5f * s;            // pre-negated bias: score = dot + lds_ww[col]
  }
  // B fragments in WMMA 16-bit B layout: lane = 16*h + n, VGPR v holds K = 8*h + 2v (+1) for v<4, 16 + 8*h + 2(v-4) (+1) for v>=4
  for (int idx = tid; idx < 2*NTILES*32; idx += THREADS) {
    int ls = idx & 31;
    int nt = (idx >> 5) & (NTILES - 1);
    int ks = idx >> 10;                      // k-step (K base = 32*ks)
    int n  = nt*16 + (ls & 15);
    int hh = ls >> 4;
    unsigned short* dst = &lds_B[idx * 16];
    #pragma unroll
    for (int v = 0; v < 8; ++v) {
      int k = ks*32 + 8*hh + 2*v + ((v >= 4) ? 8 : 0);
      dst[2*v]   = f32_to_bf16(w[(k    )*NE + n]);
      dst[2*v+1] = f32_to_bf16(w[(k + 1)*NE + n]);
    }
  }
  __syncthreads();

  float loss_acc = 0.f;

  for (int tile = blockIdx.x; tile < ROW_TILES; tile += gridDim.x) {
    const int rowbase = tile*ROWS_PER_WG + wave*16;
    const float* rp = x + (size_t)(rowbase + lm) * DIM;

    // prefetch next row tile while this one computes (global_prefetch path)
    {
      int nxt = tile + gridDim.x;
      if (nxt < ROW_TILES) {
        const float* np = x + (size_t)(nxt*ROWS_PER_WG + wave*16 + lm) * DIM;
        __builtin_prefetch(np, 0, 1);
        __builtin_prefetch(np + 32, 0, 1);
      }
    }

    // ---- load 16-row A tile: lane holds row lm, K-slices per ISA A layout ----
    float xv[32];
    v16u  au0, au1;
    float psum = 0.f;
    #pragma unroll
    for (int ks = 0; ks < 2; ++ks) {
      float4 f0 = *(const float4*)(rp + ks*32 + 8*h);
      float4 f1 = *(const float4*)(rp + ks*32 + 8*h + 4);
      float4 f2 = *(const float4*)(rp + ks*32 + 16 + 8*h);
      float4 f3 = *(const float4*)(rp + ks*32 + 16 + 8*h + 4);
      float tmp[16] = {f0.x,f0.y,f0.z,f0.w, f1.x,f1.y,f1.z,f1.w,
                       f2.x,f2.y,f2.z,f2.w, f3.x,f3.y,f3.z,f3.w};
      #pragma unroll
      for (int e = 0; e < 16; ++e) {
        xv[ks*16 + e] = tmp[e];
        unsigned short b = f32_to_bf16(tmp[e]);
        if (ks == 0) au0[e] = b; else au1[e] = b;
        psum += tmp[e]*tmp[e];
      }
    }
    v16bf a0 = __builtin_bit_cast(v16bf, au0);
    v16bf a1 = __builtin_bit_cast(v16bf, au1);
    float xx = psum + __shfl_xor(psum, 16, 32);   // full ||x_row||^2
    if (h == 0) lds_xx[wave][lm] = xx;

    // ---- scores over all 512 columns, fused argmax (rolled: best scheduling) ----
    float best[8];
    int   bidx[8];
    #pragma unroll
    for (int r = 0; r < 8; ++r) { best[r] = -3.4e38f; bidx[r] = 0; }

    #pragma unroll 1
    for (int nt = 0; nt < NTILES; ++nt) {
      v16bf b0 = __builtin_bit_cast(v16bf, *(const v16u*)&lds_B[((0*NTILES + nt)*32 + lane)*16]);
      v16bf b1 = __builtin_bit_cast(v16bf, *(const v16u*)&lds_B[((1*NTILES + nt)*32 + lane)*16]);
      v8f c = {0.f,0.f,0.f,0.f,0.f,0.f,0.f,0.f};   // SRC2 = inline constant 0
      c = __builtin_amdgcn_wmma_f32_16x16x32_bf16(false, a0, false, b0, (short)0, c, false, false);
      c = __builtin_amdgcn_wmma_f32_16x16x32_bf16(false, a1, false, b1, (short)0, c, false, false);
      const int   col = nt*16 + lm;
      const float hwv = lds_ww[col];                // -0.5*||w_col||^2 (load hidden behind WMMAs)
      #pragma unroll
      for (int r = 0; r < 8; ++r) {                 // C layout: VGPR r = row h*8+r, lane = col
        float sc = c[r] + hwv;
        if (sc > best[r]) { best[r] = sc; bidx[r] = col; }
      }
    }
    // cross-lane argmax within each 16-lane group (first-min tie-break)
    #pragma unroll
    for (int r = 0; r < 8; ++r) {
      #pragma unroll
      for (int m = 1; m < 16; m <<= 1) {
        float ov = __shfl_xor(best[r], m, 32);
        int   oi = __shfl_xor(bidx[r], m, 32);
        if (ov > best[r] || (ov == best[r] && oi < bidx[r])) { best[r] = ov; bidx[r] = oi; }
      }
    }

    // ---- bookkeeping: lanes 0-7 own rows 0-7, lanes 16-23 own rows 8-15 ----
    if (lm < 8) {
      int row = h*8 + lm;
      int j   = bidx[lm];
      atomicAdd(&lds_counts[j], 1.0f);
      // dist = ||x||^2 - 2*(x.w - 0.5||w||^2) = ||x||^2 - 2*score
      loss_acc += lds_xx[wave][row] - 2.0f * best[lm];
      lds_win[wave][row] = j;
    }
    __builtin_amdgcn_wave_barrier();   // keep LDS write->read order (same-wave DS is in-order)

    // ---- embed_sum scatter: every lane adds its 32 held x values to column j ----
    int j = lds_win[wave][lm];
    #pragma unroll
    for (int i = 0; i < 32; ++i) {
      int ks = i >> 4, e = i & 15;
      int k  = ks*32 + 8*h + e + ((e >= 8) ? 8 : 0);   // inverse of A-fragment packing
      atomicAdd(&lds_embed[k*NE + j], xv[i]);
    }
  }

  // ---- flush: loss (1 atomic/wave), counts, embed_sum ----
  #pragma unroll
  for (int m = 1; m < 32; m <<= 1) loss_acc += __shfl_xor(loss_acc, m, 32);
  if (lane == 0) atomicAdd(&out[0], loss_acc);

  __syncthreads();
  for (int i = tid; i < NE; i += THREADS)     atomicAdd(&out[OUT_CS + i], lds_counts[i]);
  for (int i = tid; i < DIM*NE; i += THREADS) atomicAdd(&out[OUT_EA + i], lds_embed[i]);
}

// ---------------- finalize: EMA update, perplexity, loss scale ----------------
__launch_bounds__(512, 1)
__global__ void vq_final(const float* __restrict__ cluster_size,
                         const float* __restrict__ embed_avg,
                         float* __restrict__ out) {
  __shared__ float s_p[512];
  __shared__ float s_n[512];
  const int t = threadIdx.x;

  float c   = out[OUT_CS + t];                       // raw counts
  float ncs = 0.99f*cluster_size[t] + (1.0f - 0.99f)*c;
  out[OUT_CS + t] = ncs;                             // new_cluster_size

  float p = c * (1.0f / (float)NCODES);
  s_p[t] = p * logf(p + 1e-10f);
  s_n[t] = ncs;
  __syncthreads();
  for (int s = 256; s > 0; s >>= 1) {
    if (t < s) { s_p[t] += s_p[t+s]; s_n[t] += s_n[t+s]; }
    __syncthreads();
  }
  const float n   = s_n[0];
  const float csj = (ncs + 1e-5f) / (n + (float)NE * 1e-5f) * n;

  for (int d = 0; d < DIM; ++d) {
    int idx = d*NE + t;
    float nea = 0.99f*embed_avg[idx] + (1.0f - 0.99f)*out[OUT_EA + idx];
    out[OUT_EA + idx] = nea;                         // new_embed_avg
    out[OUT_W  + idx] = nea / csj;                   // new_w
  }
  if (t == 0) {
    out[0] = 0.25f * out[0] / ((float)NCODES * (float)DIM);  // commitment loss
    out[1] = expf(-s_p[0]);                                   // perplexity
  }
}

// ---------------- launch ----------------
extern "C" void kernel_launch(void* const* d_in, const int* in_sizes, int n_in,
                              void* d_out, int out_size, void* d_ws, size_t ws_size,
                              hipStream_t stream) {
  (void)in_sizes; (void)n_in; (void)out_size; (void)d_ws; (void)ws_size;
  const float* input        = (const float*)d_in[0];
  const float* w            = (const float*)d_in[1];
  const float* cluster_size = (const float*)d_in[2];
  const float* embed_avg    = (const float*)d_in[3];
  float* out = (float*)d_out;

  hipLaunchKernelGGL(vq_init,  dim3(128),    dim3(256), 0, stream, out);
  hipLaunchKernelGGL(vq_main,  dim3(NUM_WG), dim3(THREADS), 0, stream, input, w, out);
  hipLaunchKernelGGL(vq_final, dim3(1),      dim3(512), 0, stream, cluster_size, embed_avg, out);
}
